// TtnnPhi3ForCausalLM_28827820491140
// MI455X (gfx1250) — compile-verified, matrix operand order checked by reading, not run on hardware
//
#include <hip/hip_runtime.h>
#include <hip/hip_bf16.h>
#include <math.h>

// ---------------- model constants ----------------
static constexpr int VOC = 32064;
static constexpr int H   = 2048;
static constexpr int I_  = 6144;
static constexpr int L_  = 2;
static constexpr int NH  = 16;
static constexpr int NKV = 4;
static constexpr int HD  = 128;
static constexpr int S_  = 1024;
static constexpr int QKVN = (NH + 2 * NKV) * HD;   // 3072
static constexpr float EPS   = 1e-5f;
static constexpr float SCALE = 0.08838834764831845f; // 1/sqrt(128)

// CDNA5 async global->LDS copy path (ASYNCcnt), if the toolchain exposes it.
#if defined(__HIP_DEVICE_COMPILE__) && \
    __has_builtin(__builtin_amdgcn_global_load_async_to_lds_b128) && \
    __has_builtin(__builtin_amdgcn_s_wait_asynccnt)
#define USE_ASYNC_COPY 1
#else
#define USE_ASYNC_COPY 0
#endif

// ---------------- WMMA types ----------------
typedef __attribute__((ext_vector_type(16))) __bf16 v16bf;
typedef __attribute__((ext_vector_type(8)))  float  v8f;
typedef __attribute__((ext_vector_type(4)))  int    v4i;

union FragAB { v16bf v; unsigned int u[8]; };

__device__ __forceinline__ unsigned short f2bf(float f) {
    unsigned int x = __float_as_uint(f);
    unsigned int r = x + 0x7fffu + ((x >> 16) & 1u);   // round-to-nearest-even
    return (unsigned short)(r >> 16);
}
__device__ __forceinline__ unsigned int pack2(float lo, float hi) {
    return (unsigned int)f2bf(lo) | ((unsigned int)f2bf(hi) << 16);
}

// =====================================================================
// Generic WMMA GEMM: C[M,N] = alpha * A[M,K] @ opB(B) (+ R)
//   ABF16: A is bf16 row-major (lda in elems); else fp32 (converted on load)
//   BTRANS: B element (k,n) read from B[n*ldb + k]
//   B fp32 (ldb), C fp32 (ldc); z-batched; B batch index = z / bZdiv
//   Block tile 128x128, K-step 64, 256 threads (8 wave32s, wave = 64x32)
//   Double-buffered LDS, 1 barrier / K-tile. bf16-A tiles use async
//   global->LDS DMA when available; B tiles staged in regs (fp32->bf16).
// =====================================================================
static constexpr int LDST = 72;   // LDS row stride in bf16 elems (144B, 16B aligned)

template<bool ABF16, bool BTRANS>
__global__ __launch_bounds__(256)
void wmma_gemm(const void* __restrict__ Av, const float* __restrict__ B,
               float* __restrict__ C, const float* __restrict__ R,
               int M, int N, int K, int lda, int ldb, int ldc,
               long aZoff, long bZoff, long cZoff, int bZdiv, float alpha)
{
    const int z = blockIdx.z;
    B += (long)(z / bZdiv) * bZoff;
    C += (long)z * cZoff;
    if (R) R += (long)z * cZoff;
    const unsigned short* Ab = (const unsigned short*)Av + (ABF16 ? (long)z * aZoff : 0);
    const float*          Af = (const float*)Av          + (ABF16 ? 0 : (long)z * aZoff);

    const int nBase = blockIdx.x * 128;
    const int mBase = blockIdx.y * 128;

    __shared__ unsigned short As[2][128][LDST];
    __shared__ unsigned short Bs[2][128][LDST];

    const int tid   = threadIdx.x;
    const int lane  = tid & 31;
    const int wid   = tid >> 5;
    const int wm    = (wid >> 2) * 64;         // wave M base within tile
    const int wn    = (wid & 3) * 32;          // wave N base within tile
    const int half8 = (lane >> 4) * 8;
    const int rc    = lane & 15;

    v8f acc[4][2];
    #pragma unroll
    for (int mt = 0; mt < 4; ++mt)
        #pragma unroll
        for (int nt = 0; nt < 2; ++nt)
            acc[mt][nt] = (v8f)(0.0f);

    unsigned int aReg[16];
    unsigned int bReg[16];

    // ---- load one 64-wide K tile (A: async DMA or reg-staged; B: reg-staged) ----
    auto loadTile = [&](int dstBuf, int kb) {
        if (ABF16) {
#if USE_ASYNC_COPY
            #pragma unroll
            for (int i = 0; i < 2; ++i) {
                int slot = tid + i * 256;          // 512 B128 slots
                int row  = slot >> 2;
                int k16  = (slot & 3) * 16;
                const unsigned short* gsrc = &Ab[(size_t)(mBase + row) * lda + kb + k16];
                unsigned int loff = (unsigned int)(unsigned long long)&As[dstBuf][row][k16];
                __builtin_amdgcn_global_load_async_to_lds_b128(
                    (__attribute__((address_space(1))) v4i*)(unsigned long long)gsrc,
                    (__attribute__((address_space(3))) v4i*)loff,
                    0, 0);
            }
#else
            #pragma unroll
            for (int i = 0; i < 2; ++i) {
                int slot = tid + i * 256;
                int row  = slot >> 2;
                int k16  = (slot & 3) * 16;
                uint4 v = *(const uint4*)&Ab[(size_t)(mBase + row) * lda + kb + k16];
                aReg[i * 4 + 0] = v.x; aReg[i * 4 + 1] = v.y;
                aReg[i * 4 + 2] = v.z; aReg[i * 4 + 3] = v.w;
            }
#endif
        } else {
            #pragma unroll
            for (int i = 0; i < 8; ++i) {
                int slot = tid + i * 256;          // 2048 float4 slots
                int row  = slot >> 4;
                int k4   = (slot & 15) * 4;
                float4 a = *(const float4*)&Af[(size_t)(mBase + row) * lda + kb + k4];
                aReg[2 * i]     = pack2(a.x, a.y);
                aReg[2 * i + 1] = pack2(a.z, a.w);
            }
        }
        if (BTRANS) {
            #pragma unroll
            for (int i = 0; i < 8; ++i) {
                int slot = tid + i * 256;
                int n    = slot >> 4;
                int k4   = (slot & 15) * 4;
                float4 b = make_float4(0.f, 0.f, 0.f, 0.f);
                if (nBase + n < N)
                    b = *(const float4*)&B[(size_t)(nBase + n) * ldb + kb + k4];
                bReg[2 * i]     = pack2(b.x, b.y);
                bReg[2 * i + 1] = pack2(b.z, b.w);
            }
        } else {
            #pragma unroll
            for (int i = 0; i < 8; ++i) {
                int slot = tid + i * 256;
                int krow = slot >> 5;
                int c4   = (slot & 31) * 4;
                const float* bp = &B[(size_t)(kb + krow) * ldb + nBase + c4];
                float4 b;
                if (nBase + c4 + 3 < N) {
                    b = *(const float4*)bp;
                } else {
                    b.x = (nBase + c4 + 0 < N) ? bp[0] : 0.f;
                    b.y = (nBase + c4 + 1 < N) ? bp[1] : 0.f;
                    b.z = (nBase + c4 + 2 < N) ? bp[2] : 0.f;
                    b.w = (nBase + c4 + 3 < N) ? bp[3] : 0.f;
                }
                bReg[2 * i]     = pack2(b.x, b.y);
                bReg[2 * i + 1] = pack2(b.z, b.w);
            }
        }
        // prefetch next tile (global_prefetch_b8)
        if (kb + 64 < K) {
            if (ABF16)
                __builtin_prefetch(&Ab[(size_t)(mBase + (tid & 127)) * lda + kb + 64], 0, 1);
            else
                __builtin_prefetch(&Af[(size_t)(mBase + (tid & 127)) * lda + kb + 64], 0, 1);
            if (!BTRANS)
                __builtin_prefetch(&B[(size_t)(kb + 64 + (tid & 63)) * ldb + nBase], 0, 1);
        }
    };

    // ---- commit staged registers to LDS buffer `buf` ----
    auto storeTile = [&](int buf) {
        if (ABF16) {
#if !USE_ASYNC_COPY
            #pragma unroll
            for (int i = 0; i < 2; ++i) {
                int slot = tid + i * 256;
                int row  = slot >> 2;
                int k16  = (slot & 3) * 16;
                uint4 v = make_uint4(aReg[i * 4 + 0], aReg[i * 4 + 1],
                                     aReg[i * 4 + 2], aReg[i * 4 + 3]);
                *(uint4*)&As[buf][row][k16] = v;
            }
#endif
        } else {
            #pragma unroll
            for (int i = 0; i < 8; ++i) {
                int slot = tid + i * 256;
                int row  = slot >> 4;
                int k4   = (slot & 15) * 4;
                *(unsigned int*)&As[buf][row][k4]     = aReg[2 * i];
                *(unsigned int*)&As[buf][row][k4 + 2] = aReg[2 * i + 1];
            }
        }
        if (BTRANS) {
            #pragma unroll
            for (int i = 0; i < 8; ++i) {
                int slot = tid + i * 256;
                int n    = slot >> 4;
                int k4   = (slot & 15) * 4;
                *(unsigned int*)&Bs[buf][n][k4]     = bReg[2 * i];
                *(unsigned int*)&Bs[buf][n][k4 + 2] = bReg[2 * i + 1];
            }
        } else {
            #pragma unroll
            for (int i = 0; i < 8; ++i) {
                int slot = tid + i * 256;
                int krow = slot >> 5;
                int c4   = (slot & 31) * 4;
                Bs[buf][c4 + 0][krow] = (unsigned short)(bReg[2 * i]);
                Bs[buf][c4 + 1][krow] = (unsigned short)(bReg[2 * i] >> 16);
                Bs[buf][c4 + 2][krow] = (unsigned short)(bReg[2 * i + 1]);
                Bs[buf][c4 + 3][krow] = (unsigned short)(bReg[2 * i + 1] >> 16);
            }
        }
    };

    // wait for this wave's outstanding async LDS deposits before the barrier
    auto asyncWait = [&]() {
#if USE_ASYNC_COPY
        if (ABF16) __builtin_amdgcn_s_wait_asynccnt(0);
#endif
    };

    // ---- 16 WMMAs on LDS buffer `buf` ----
    auto compute = [&](int buf) {
        #pragma unroll
        for (int ks = 0; ks < 64; ks += 32) {
            FragAB af[4], bfc[2];
            #pragma unroll
            for (int mt = 0; mt < 4; ++mt) {
                const int row = wm + mt * 16 + rc;
                #pragma unroll
                for (int v = 0; v < 8; ++v) {
                    int kk = ks + 2 * v + ((v >= 4) ? 8 : 0) + half8;
                    af[mt].u[v] = *(const unsigned int*)&As[buf][row][kk];
                }
            }
            #pragma unroll
            for (int nt = 0; nt < 2; ++nt) {
                const int col = wn + nt * 16 + rc;
                #pragma unroll
                for (int v = 0; v < 8; ++v) {
                    int kk = ks + 2 * v + ((v >= 4) ? 8 : 0) + half8;
                    bfc[nt].u[v] = *(const unsigned int*)&Bs[buf][col][kk];
                }
            }
            #pragma unroll
            for (int mt = 0; mt < 4; ++mt)
                #pragma unroll
                for (int nt = 0; nt < 2; ++nt)
                    acc[mt][nt] = __builtin_amdgcn_wmma_f32_16x16x32_bf16(
                        false, af[mt].v, false, bfc[nt].v,
                        (short)0, acc[mt][nt], false, false);
        }
    };

    // ---- software pipeline: 1 barrier per K-tile ----
    const int nTiles = K >> 6;
    loadTile(0, 0);
    storeTile(0);
    asyncWait();
    __syncthreads();
    for (int t = 0; t < nTiles; ++t) {
        const int cur = t & 1;
        const bool hasNext = (t + 1 < nTiles);
        if (hasNext) loadTile(1 - cur, (t + 1) << 6);   // VMEM/DMA in flight during WMMA
        compute(cur);
        if (hasNext) { storeTile(1 - cur); asyncWait(); }
        __syncthreads();
    }

    // -------- epilogue: C = alpha*acc (+R) --------
    const int rbase = (lane >> 4) * 8;
    const int cc    = lane & 15;
    #pragma unroll
    for (int mt = 0; mt < 4; ++mt) {
        #pragma unroll
        for (int nt = 0; nt < 2; ++nt) {
            const int col = nBase + wn + nt * 16 + cc;
            if (col < N) {
                #pragma unroll
                for (int r = 0; r < 8; ++r) {
                    const int row = mBase + wm + mt * 16 + rbase + r;
                    if (row < M) {
                        size_t idx = (size_t)row * ldc + col;
                        float v = acc[mt][nt][r] * alpha;
                        if (R) v += R[idx];
                        C[idx] = v;
                    }
                }
            }
        }
    }
}

// =====================================================================
// Elementwise / reduction kernels
// =====================================================================
__global__ void embed_kernel(const int* __restrict__ ids,
                             const float* __restrict__ emb,
                             float* __restrict__ x)
{
    int idx = blockIdx.x * blockDim.x + threadIdx.x;
    if (idx >= S_ * H) return;
    int s = idx >> 11;          // /H
    int h = idx & (H - 1);
    x[idx] = emb[(size_t)ids[s] * H + h];
}

// fp32 in -> bf16 normalized out
__global__ void rmsnorm_kernel(const float* __restrict__ x,
                               const float* __restrict__ w,
                               unsigned short* __restrict__ out, int ncols)
{
    const int row = blockIdx.x;
    const float* xr = x + (size_t)row * ncols;
    float ss = 0.f;
    for (int i = threadIdx.x; i < ncols; i += blockDim.x) {
        float v = xr[i]; ss += v * v;
    }
    __shared__ float red[256];
    red[threadIdx.x] = ss;
    __syncthreads();
    for (int s = 128; s > 0; s >>= 1) {
        if (threadIdx.x < s) red[threadIdx.x] += red[threadIdx.x + s];
        __syncthreads();
    }
    float rs = rsqrtf(red[0] / (float)ncols + EPS);
    for (int i = threadIdx.x; i < ncols; i += blockDim.x)
        out[(size_t)row * ncols + i] = f2bf(xr[i] * w[i] * rs);
}

__global__ void rope_kernel(float* __restrict__ qkv)
{
    // heads 0..15 = Q, 16..19 = K (contiguous at head*HD within qkv row)
    int idx = blockIdx.x * blockDim.x + threadIdx.x;
    const int total = S_ * (NH + NKV) * (HD / 2);
    if (idx >= total) return;
    int d    = idx & 63;
    int t    = idx >> 6;
    int head = t % (NH + NKV);
    int s    = t / (NH + NKV);
    float* base = qkv + (size_t)s * QKVN + head * HD;
    float invf = __expf(-__logf(10000.f) * ((float)(2 * d)) * (1.f / (float)HD));
    float ang = (float)s * invf;
    float c, sn;
    __sincosf(ang, &sn, &c);
    float x1 = base[d], x2 = base[d + 64];
    base[d]      = x1 * c - x2 * sn;
    base[d + 64] = x2 * c + x1 * sn;
}

// scores fp32 in -> causal softmax -> bf16 probabilities out
__global__ void softmax_kernel(const float* __restrict__ scores,
                               unsigned short* __restrict__ p, float scale)
{
    const int q = blockIdx.x;
    const int h = blockIdx.y;
    const size_t off = ((size_t)h * S_ + q) * S_;
    const float* row = scores + off;
    __shared__ float red[256];
    const int tid = threadIdx.x;

    float m = -1e30f;
    for (int i = tid; i < S_; i += 256)
        if (i <= q) m = fmaxf(m, row[i] * scale);
    red[tid] = m;
    __syncthreads();
    for (int s = 128; s > 0; s >>= 1) {
        if (tid < s) red[tid] = fmaxf(red[tid], red[tid + s]);
        __syncthreads();
    }
    const float M = red[0];
    __syncthreads();

    float sum = 0.f;
    for (int i = tid; i < S_; i += 256)
        if (i <= q) sum += __expf(row[i] * scale - M);
    red[tid] = sum;
    __syncthreads();
    for (int s = 128; s > 0; s >>= 1) {
        if (tid < s) red[tid] += red[tid + s];
        __syncthreads();
    }
    const float inv = 1.f / red[0];
    for (int i = tid; i < S_; i += 256)
        p[off + i] = f2bf((i <= q) ? __expf(row[i] * scale - M) * inv : 0.f);
}

__global__ void silu_mul_kernel(const float* __restrict__ gu,
                                unsigned short* __restrict__ out)
{
    int idx = blockIdx.x * blockDim.x + threadIdx.x;
    if (idx >= S_ * I_) return;
    int s = idx / I_;
    int i = idx - s * I_;
    float g = gu[(size_t)s * (2 * I_) + i];
    float u = gu[(size_t)s * (2 * I_) + I_ + i];
    out[idx] = f2bf((g / (1.f + __expf(-g))) * u);
}

// =====================================================================
// Host launch
// =====================================================================
enum GemmMode { GE_BF16A = 0, GE_F32A = 1, GE_F32A_BT = 2 };

static inline void gemm(GemmMode mode, const void* A, const float* B, float* C,
                        const float* R, int M, int N, int K,
                        int lda, int ldb, int ldc,
                        long aZ, long bZ, long cZ, int bDiv,
                        float alpha, int Z, hipStream_t st)
{
    dim3 grid((N + 127) / 128, (M + 127) / 128, Z), blk(256);
    switch (mode) {
    case GE_BF16A:
        hipLaunchKernelGGL((wmma_gemm<true, false>), grid, blk, 0, st,
                           A, B, C, R, M, N, K, lda, ldb, ldc, aZ, bZ, cZ, bDiv, alpha);
        break;
    case GE_F32A:
        hipLaunchKernelGGL((wmma_gemm<false, false>), grid, blk, 0, st,
                           A, B, C, R, M, N, K, lda, ldb, ldc, aZ, bZ, cZ, bDiv, alpha);
        break;
    case GE_F32A_BT:
        hipLaunchKernelGGL((wmma_gemm<false, true>), grid, blk, 0, st,
                           A, B, C, R, M, N, K, lda, ldb, ldc, aZ, bZ, cZ, bDiv, alpha);
        break;
    }
}

extern "C" void kernel_launch(void* const* d_in, const int* in_sizes, int n_in,
                              void* d_out, int out_size, void* d_ws, size_t ws_size,
                              hipStream_t stream)
{
    const int*   ids       = (const int*)  d_in[0];
    const float* emb       = (const float*)d_in[1];
    const float* qkv_w     = (const float*)d_in[2];
    const float* o_w       = (const float*)d_in[3];
    const float* gate_up_w = (const float*)d_in[4];
    const float* down_w    = (const float*)d_in[5];
    const float* ln1_w     = (const float*)d_in[6];
    const float* ln2_w     = (const float*)d_in[7];
    const float* fln_w     = (const float*)d_in[8];
    const float* lm_w      = (const float*)d_in[9];
    float* out = (float*)d_out;

    // workspace layout (256B aligned)
    char* p = (char*)d_ws;
    auto alloc = [&](size_t bytes) {
        char* r = p;
        p += (bytes + 255) & ~(size_t)255;
        return r;
    };
    float*          x      = (float*)alloc((size_t)S_ * H * 4);          // residual
    float*          qkv    = (float*)alloc((size_t)S_ * QKVN * 4);
    float*          scores = (float*)alloc((size_t)NH * S_ * S_ * 4);
    float*          attn   = (float*)alloc((size_t)S_ * NH * HD * 4);
    float*          gu     = (float*)alloc((size_t)S_ * 2 * I_ * 4);
    unsigned short* hb     = (unsigned short*)alloc((size_t)S_ * H * 2);     // bf16 activations
    unsigned short* sb     = (unsigned short*)alloc((size_t)S_ * I_ * 2);    // bf16 silu*up
    unsigned short* pbuf   = (unsigned short*)alloc((size_t)NH * S_ * S_ * 2); // bf16 probs
    (void)ws_size; (void)n_in; (void)in_sizes; (void)out_size;

    embed_kernel<<<(S_ * H + 255) / 256, 256, 0, stream>>>(ids, emb, x);

    for (int l = 0; l < L_; ++l) {
        rmsnorm_kernel<<<S_, 256, 0, stream>>>(x, ln1_w + (size_t)l * H, hb, H);
        // qkv = h @ qkv_w[l]
        gemm(GE_BF16A, hb, qkv_w + (size_t)l * H * QKVN, qkv, nullptr,
             S_, QKVN, H, H, QKVN, QKVN, 0, 0, 0, 1, 1.f, 1, stream);
        rope_kernel<<<(S_ * (NH + NKV) * 64 + 255) / 256, 256, 0, stream>>>(qkv);
        // scores[h] = Q_h @ K_{h/4}^T  (A fp32, B transposed, z over heads)
        gemm(GE_F32A_BT, qkv, qkv + NH * HD, scores, nullptr,
             S_, S_, HD, QKVN, QKVN, S_,
             (long)HD, (long)HD, (long)S_ * S_, NH / NKV, 1.f, NH, stream);
        softmax_kernel<<<dim3(S_, NH), 256, 0, stream>>>(scores, pbuf, SCALE);
        // attn[h] = P_h @ V_{h/4}   (A = bf16 probabilities)
        gemm(GE_BF16A, pbuf, qkv + (NH + NKV) * HD, attn, nullptr,
             S_, HD, S_, S_, QKVN, NH * HD,
             (long)S_ * S_, (long)HD, (long)HD, NH / NKV, 1.f, NH, stream);
        // x = x + attn @ o_w[l]
        gemm(GE_F32A, attn, o_w + (size_t)l * (NH * HD) * H, x, x,
             S_, H, NH * HD, NH * HD, H, H, 0, 0, 0, 1, 1.f, 1, stream);
        rmsnorm_kernel<<<S_, 256, 0, stream>>>(x, ln2_w + (size_t)l * H, hb, H);
        // gu = h @ gate_up_w[l]
        gemm(GE_BF16A, hb, gate_up_w + (size_t)l * H * (2 * I_), gu, nullptr,
             S_, 2 * I_, H, H, 2 * I_, 2 * I_, 0, 0, 0, 1, 1.f, 1, stream);
        silu_mul_kernel<<<(S_ * I_ + 255) / 256, 256, 0, stream>>>(gu, sb);
        // x = x + sbuf @ down_w[l]
        gemm(GE_BF16A, sb, down_w + (size_t)l * I_ * H, x, x,
             S_, H, I_, I_, H, H, 0, 0, 0, 1, 1.f, 1, stream);
    }

    rmsnorm_kernel<<<S_, 256, 0, stream>>>(x, fln_w, hb, H);
    gemm(GE_BF16A, hb, lm_w, out, nullptr,
         S_, VOC, H, H, VOC, VOC, 0, 0, 0, 1, 1.f, 1, stream);
}
